// Model_77300821393510
// MI455X (gfx1250) — matrix operand order repaired and checked by reading, not compiled, not run on hardware
//
#include <hip/hip_runtime.h>
#include <math.h>

typedef _Float16 half16 __attribute__((ext_vector_type(16)));
typedef _Float16 half8  __attribute__((ext_vector_type(8)));
typedef _Float16 half2v __attribute__((ext_vector_type(2)));
typedef float    float8 __attribute__((ext_vector_type(8)));

#define DM   256      // d_model
#define NH   8        // heads
#define HE   32       // head dim
#define SL   1024     // seq len (enc & dec)
#define BB   8        // batch
#define DFF  1024
#define CIN  7
#define FIN  7175     // 7*(1+1024)
#define DHID 128
#define LN_EPS 1e-5f
#define NEG_BIG (-1.0e30f)

// build A-layout fragment (16x32 f16) from two contiguous 8-half chunks
__device__ __forceinline__ half16 make_afrag(const _Float16* base /*row start*/, int kg)
{
    half8 lo = *reinterpret_cast<const half8*>(base + 2 * kg);
    half8 hi = *reinterpret_cast<const half8*>(base + 2 * kg + 8);
    return __builtin_shufflevector(lo, hi, 0, 1, 2, 3, 4, 5, 6, 7,
                                   8, 9, 10, 11, 12, 13, 14, 15);
}

// ---------------------------------------------------------------------------
// mean / unbiased std over SEQ per (b, c)
// ---------------------------------------------------------------------------
__global__ __launch_bounds__(256) void stats_kernel(
    const float* __restrict__ x, float* __restrict__ mu, float* __restrict__ sig)
{
    int bc = blockIdx.x;              // b*CIN + c
    int b = bc / CIN, c = bc % CIN;
    int t = threadIdx.x;
    float s = 0.f, sq = 0.f;
    for (int l = t; l < SL; l += 256) {
        float v = x[((size_t)b * SL + l) * CIN + c];
        s += v; sq += v * v;
    }
    __shared__ float r0[256], r1[256];
    r0[t] = s; r1[t] = sq; __syncthreads();
    for (int o = 128; o > 0; o >>= 1) {
        if (t < o) { r0[t] += r0[t + o]; r1[t] += r1[t + o]; }
        __syncthreads();
    }
    if (t == 0) {
        float m = r0[0] / (float)SL;
        float var = (r1[0] - (float)SL * m * m) / (float)(SL - 1);
        mu[bc] = m;
        sig[bc] = sqrtf(fmaxf(var, 0.f)) + 1e-5f;
    }
}

// ---------------------------------------------------------------------------
// factor hidden: out[b*128+j] = concat(head, x_enc)[b,:] . W1[:,j] + b1[j]
// ---------------------------------------------------------------------------
__global__ __launch_bounds__(256) void factor_hidden_kernel(
    const float* __restrict__ x_enc, const float* __restrict__ head7,
    const float* __restrict__ W1, const float* __restrict__ b1,
    float* __restrict__ outh)
{
    int j = blockIdx.x, b = blockIdx.y, t = threadIdx.x;
    float s = 0.f;
    for (int i = t; i < FIN; i += 256) {
        float v = (i < CIN) ? head7[b * CIN + i]
                            : x_enc[(size_t)b * (SL * CIN) + (i - CIN)];
        s += v * W1[(size_t)i * DHID + j];
    }
    __shared__ float r0[256];
    r0[t] = s; __syncthreads();
    for (int o = 128; o > 0; o >>= 1) { if (t < o) r0[t] += r0[t + o]; __syncthreads(); }
    if (t == 0) outh[b * DHID + j] = r0[0] + b1[j];
}

// tau[b] = exp((tauh[b,:].W2 + b2) / sqrt(7168))
__global__ __launch_bounds__(128) void tau_kernel(
    const float* __restrict__ tauh, const float* __restrict__ W2,
    const float* __restrict__ b2, float* __restrict__ tau)
{
    int b = blockIdx.x, t = threadIdx.x;
    __shared__ float r0[128];
    r0[t] = tauh[b * DHID + t] * W2[t]; __syncthreads();
    for (int o = 64; o > 0; o >>= 1) { if (t < o) r0[t] += r0[t + o]; __syncthreads(); }
    if (t == 0) tau[b] = __expf((r0[0] + b2[0]) / sqrtf((float)(CIN * SL)));
}

// delta[b,s] = delh[b,:].W2[:,s] + b2[s]
__global__ __launch_bounds__(256) void delta_kernel(
    const float* __restrict__ delh, const float* __restrict__ W2,
    const float* __restrict__ b2, float* __restrict__ delta)
{
    int idx = blockIdx.x * 256 + threadIdx.x;   // B*SL total
    int b = idx >> 10, s = idx & (SL - 1);
    float acc = b2[s];
    #pragma unroll 4
    for (int j = 0; j < DHID; ++j)
        acc += delh[b * DHID + j] * W2[(size_t)j * SL + s];
    delta[idx] = acc;
}

// ---------------------------------------------------------------------------
// data embedding: circular conv1d(k=3) + positional + temporal
// ---------------------------------------------------------------------------
__global__ __launch_bounds__(256) void embed_kernel(
    const float* __restrict__ x, const float* __restrict__ xmark,
    const float* __restrict__ conv, const float* __restrict__ tempW,
    const float* __restrict__ mu, const float* __restrict__ sig,
    float* __restrict__ out, int standardize)
{
    size_t idx = (size_t)blockIdx.x * 256 + threadIdx.x;    // B*SL*DM
    int d = idx & (DM - 1);
    int l = (idx >> 8) & (SL - 1);
    int b = (int)(idx >> 18);
    float acc = 0.f;
    #pragma unroll
    for (int k = 0; k < 3; ++k) {
        int ls = l + k - 1;
        ls = (ls < 0) ? SL - 1 : (ls >= SL ? 0 : ls);
        #pragma unroll
        for (int c = 0; c < CIN; ++c) {
            float xv = x[((size_t)b * SL + ls) * CIN + c];
            if (standardize) xv = (xv - mu[b * CIN + c]) / sig[b * CIN + c];
            acc += xv * conv[(k * CIN + c) * DM + d];
        }
    }
    int j = d >> 1;
    float div = __expf(-logf(10000.f) * (float)(2 * j) / (float)DM);
    float ang = (float)l * div;
    acc += (d & 1) ? cosf(ang) : sinf(ang);
    #pragma unroll
    for (int m = 0; m < 4; ++m)
        acc += xmark[((size_t)b * SL + l) * 4 + m] * tempW[m * DM + d];
    out[idx] = acc;
}

// ---------------------------------------------------------------------------
// Generic WMMA GEMM: C[M,N] = act(A[M,K] @ W[K,N] + bias)
// block tile 64x256, 8 waves (2x4), wave tile 32x64, K step 32.
// A staged [m][k] (f16), W staged transposed [n][k] (f16) so every fragment
// is a contiguous 16/32-byte LDS load. requires M%64==0, N%256==0, K%32==0.
// ---------------------------------------------------------------------------
__global__ __launch_bounds__(256) void gemm_wmma_kernel(
    const float* __restrict__ A, const float* __restrict__ W,
    const float* __restrict__ bias, float* __restrict__ C,
    int M, int N, int K, int act)
{
    __shared__ __align__(32) _Float16 As[64 * 32];    // [m][k]
    __shared__ __align__(32) _Float16 Bt[256 * 32];   // [n][k]  (transposed)
    const int tid = threadIdx.x;
    const int lane = tid & 31, wave = tid >> 5;
    const int wm = wave >> 2;        // 0..1  (M dir, 32 rows)
    const int wn = wave & 3;         // 0..3  (N dir, 64 cols)
    const int m_blk = blockIdx.y * 64;
    const int n_blk = blockIdx.x * 256;
    const int hl = lane & 15, lh = lane >> 4;

    float8 acc[2][4];
    #pragma unroll
    for (int a = 0; a < 2; ++a)
        #pragma unroll
        for (int bq = 0; bq < 4; ++bq) { float8 z = {}; acc[a][bq] = z; }

    for (int k0 = 0; k0 < K; k0 += 32) {
        __syncthreads();
        // stage A tile 64x32 as f16 pairs (coalesced along k)
        for (int i = tid; i < 64 * 16; i += 256) {
            int r = i >> 4, cp = (i & 15) << 1;
            const float* src = &A[(size_t)(m_blk + r) * K + (k0 + cp)];
            half2v hv; hv[0] = (_Float16)src[0]; hv[1] = (_Float16)src[1];
            *reinterpret_cast<half2v*>(&As[r * 32 + cp]) = hv;
        }
        // stage W tile 32x256 transposed -> Bt[n][k] (coalesced along n)
        for (int i = tid; i < 256 * 16; i += 256) {
            int c = i & 255, rp = (i >> 8) << 1;
            half2v hv;
            hv[0] = (_Float16)W[(size_t)(k0 + rp) * N + (n_blk + c)];
            hv[1] = (_Float16)W[(size_t)(k0 + rp + 1) * N + (n_blk + c)];
            *reinterpret_cast<half2v*>(&Bt[c * 32 + rp]) = hv;
        }
        if (k0 + 32 < K) {  // warm caches for next tiles
            __builtin_prefetch(&A[(size_t)(m_blk + (tid >> 2)) * K + k0 + 32], 0, 3);
            __builtin_prefetch(&W[(size_t)(k0 + 32 + (tid >> 5)) * N + n_blk + ((tid & 31) << 3)], 0, 3);
        }
        __syncthreads();

        // B fragments: contiguous 32-byte LDS loads
        half16 bf[4];
        #pragma unroll
        for (int nt = 0; nt < 4; ++nt) {
            int n = wn * 64 + nt * 16 + hl;
            bf[nt] = *reinterpret_cast<const half16*>(&Bt[n * 32 + lh * 16]);
        }
        #pragma unroll
        for (int mt = 0; mt < 2; ++mt) {
            int m = wm * 32 + mt * 16 + hl;
            half16 af = make_afrag(&As[m * 32], lh * 8);
            #pragma unroll
            for (int nt = 0; nt < 4; ++nt)
                acc[mt][nt] = __builtin_amdgcn_wmma_f32_16x16x32_f16(
                    false, af, false, bf[nt], (short)0, acc[mt][nt], false, false);
        }
    }

    // epilogue: bias + optional exact GELU
    #pragma unroll
    for (int mt = 0; mt < 2; ++mt)
        #pragma unroll
        for (int nt = 0; nt < 4; ++nt) {
            int n = n_blk + wn * 64 + nt * 16 + hl;
            #pragma unroll
            for (int i = 0; i < 8; ++i) {
                int m = m_blk + wm * 32 + mt * 16 + i + lh * 8;
                float v = acc[mt][nt][i];
                if (bias) v += bias[n];
                if (act == 1) v = 0.5f * v * (1.0f + erff(v * 0.70710678118654752f));
                C[(size_t)m * N + n] = v;
            }
        }
}

// ---------------------------------------------------------------------------
// Fused flash attention with WMMA. grid (SL/64, NH, B), 128 threads (4 waves),
// each wave owns 16 query rows. logits = (tau*qk [+ delta]) / sqrt(E).
// K staged [key][e] (score B-frags contiguous), V staged transposed [e][key]
// (PV B-frags contiguous).
// ---------------------------------------------------------------------------
__global__ __launch_bounds__(128) void attn_wmma_kernel(
    const float* __restrict__ Q, const float* __restrict__ Kb,
    const float* __restrict__ Vb, float* __restrict__ O,
    const float* __restrict__ tau_p, const float* __restrict__ delta_p,
    int causal, int apply_delta)
{
    __shared__ __align__(32) _Float16 Ks[32 * 32];      // [key][e]
    __shared__ __align__(32) _Float16 Vt[32 * 32];      // [e][key]
    __shared__ __align__(32) _Float16 Ps[4][16 * 32];   // per-wave P [row][key]
    const int tid = threadIdx.x, lane = tid & 31, wave = tid >> 5;
    const int b = blockIdx.z, h = blockIdx.y;
    const int row0 = blockIdx.x * 64 + wave * 16;
    const float tau = tau_p[b];
    const float inv_sqrtE = 0.17677669529663687f;   // 1/sqrt(32)
    const int hl = lane & 15, lh = lane >> 4;

    // Q fragment (A-layout 16x32)
    half16 qf;
    {
        int kg = lh * 8;
        const float* qrow = Q + ((size_t)(b * SL + row0 + hl) * DM) + h * HE;
        #pragma unroll
        for (int p = 0; p < 8; ++p) {
            qf[2 * p]     = (_Float16)qrow[2 * kg + 2 * p];
            qf[2 * p + 1] = (_Float16)qrow[2 * kg + 2 * p + 1];
        }
    }

    float8 o0 = {}, o1 = {};
    float mrow[8], lrow[8];
    #pragma unroll
    for (int i = 0; i < 8; ++i) { mrow[i] = NEG_BIG; lrow[i] = 0.f; }

    const int s_end = causal ? (blockIdx.x * 64 + 64) : SL;
    for (int s0 = 0; s0 < s_end; s0 += 32) {
        __syncthreads();
        // stage K chunk [key][e] (coalesced along e, packed b32 stores)
        for (int i = tid; i < 512; i += 128) {
            int s = i >> 4, ep = (i & 15) << 1;
            size_t off = ((size_t)(b * SL + s0 + s) * DM) + h * HE + ep;
            half2v kv; kv[0] = (_Float16)Kb[off]; kv[1] = (_Float16)Kb[off + 1];
            *reinterpret_cast<half2v*>(&Ks[s * 32 + ep]) = kv;
        }
        // stage V chunk transposed [e][key]
        for (int i = tid; i < 512; i += 128) {
            int e = i >> 4, sp = (i & 15) << 1;
            half2v vv;
            vv[0] = (_Float16)Vb[((size_t)(b * SL + s0 + sp) * DM) + h * HE + e];
            vv[1] = (_Float16)Vb[((size_t)(b * SL + s0 + sp + 1) * DM) + h * HE + e];
            *reinterpret_cast<half2v*>(&Vt[e * 32 + sp]) = vv;
        }
        __syncthreads();

        // scores: Q(16x32) x K^T(32x16) twice; B-frag = contiguous Ks row
        float8 sc[2];
        #pragma unroll
        for (int nt = 0; nt < 2; ++nt) {
            int n = nt * 16 + hl;                // key within chunk
            half16 bf = *reinterpret_cast<const half16*>(&Ks[n * 32 + lh * 16]);
            float8 z = {};
            sc[nt] = __builtin_amdgcn_wmma_f32_16x16x32_f16(
                false, qf, false, bf, (short)0, z, false, false);
        }

        int key0 = s0 + hl;
        int key1 = s0 + 16 + hl;
        float d0 = apply_delta ? delta_p[b * SL + key0] : 0.f;
        float d1 = apply_delta ? delta_p[b * SL + key1] : 0.f;

        #pragma unroll
        for (int i = 0; i < 8; ++i) {
            int r = row0 + i + lh * 8;
            float l0 = (tau * sc[0][i] + d0) * inv_sqrtE;
            float l1 = (tau * sc[1][i] + d1) * inv_sqrtE;
            if (causal) {
                if (key0 > r) l0 = NEG_BIG;
                if (key1 > r) l1 = NEG_BIG;
            }
            float cm = fmaxf(l0, l1);
            #pragma unroll
            for (int off = 1; off < 16; off <<= 1)
                cm = fmaxf(cm, __shfl_xor(cm, off, 32));
            float nm = fmaxf(mrow[i], cm);
            float corr = __expf(mrow[i] - nm);
            float p0 = __expf(l0 - nm);
            float p1 = __expf(l1 - nm);
            float ps = p0 + p1;
            #pragma unroll
            for (int off = 1; off < 16; off <<= 1)
                ps += __shfl_xor(ps, off, 32);
            lrow[i] = lrow[i] * corr + ps;
            mrow[i] = nm;
            o0[i] *= corr; o1[i] *= corr;
            int pr = i + lh * 8;                 // C-layout row within tile
            Ps[wave][pr * 32 + hl]      = (_Float16)p0;
            Ps[wave][pr * 32 + 16 + hl] = (_Float16)p1;
        }
        __syncthreads();

        // O += P(16x32 keys) x V(32x32): P as A-frag, V-frag = contiguous Vt row
        half16 pf = make_afrag(&Ps[wave][hl * 32], lh * 8);
        #pragma unroll
        for (int nt = 0; nt < 2; ++nt) {
            int e = nt * 16 + hl;
            half16 vf = *reinterpret_cast<const half16*>(&Vt[e * 32 + lh * 16]);
            if (nt == 0)
                o0 = __builtin_amdgcn_wmma_f32_16x16x32_f16(false, pf, false, vf, (short)0, o0, false, false);
            else
                o1 = __builtin_amdgcn_wmma_f32_16x16x32_f16(false, pf, false, vf, (short)0, o1, false, false);
        }
    }

    // normalize + write (C-layout)
    #pragma unroll
    for (int i = 0; i < 8; ++i) {
        int r = row0 + i + lh * 8;
        float inv = 1.0f / lrow[i];
        size_t base = ((size_t)(b * SL + r) * DM) + h * HE;
        O[base + hl]      = o0[i] * inv;
        O[base + 16 + hl] = o1[i] * inv;
    }
}

// ---------------------------------------------------------------------------
// out = LN(x [+ y]) * g + b ; one block per row of 256
// ---------------------------------------------------------------------------
__global__ __launch_bounds__(256) void add_ln_kernel(
    const float* __restrict__ X, const float* __restrict__ Y,
    const float* __restrict__ g, const float* __restrict__ bta,
    float* __restrict__ Out)
{
    size_t row = blockIdx.x;
    int t = threadIdx.x;
    float v = X[row * DM + t];
    if (Y) v += Y[row * DM + t];
    __shared__ float sh[256];
    sh[t] = v; __syncthreads();
    for (int o = 128; o > 0; o >>= 1) { if (t < o) sh[t] += sh[t + o]; __syncthreads(); }
    float mean = sh[0] / (float)DM;
    __syncthreads();
    float dv = v - mean;
    sh[t] = dv * dv; __syncthreads();
    for (int o = 128; o > 0; o >>= 1) { if (t < o) sh[t] += sh[t + o]; __syncthreads(); }
    float var = sh[0] / (float)DM;
    Out[row * DM + t] = g[t] * dv * rsqrtf(var + LN_EPS) + bta[t];
}

// ---------------------------------------------------------------------------
// final: out[b,lp,c] = sigma[b,c] * (dec[b,512+lp,:] . projW[:,c] + projb[c] + mu[b,c])
// ---------------------------------------------------------------------------
__global__ __launch_bounds__(256) void output_kernel(
    const float* __restrict__ dec, const float* __restrict__ projW,
    const float* __restrict__ projb, const float* __restrict__ mu,
    const float* __restrict__ sig, float* __restrict__ out)
{
    int idx = blockIdx.x * 256 + threadIdx.x;
    if (idx >= BB * 512 * CIN) return;
    int c = idx % CIN;
    int lp = (idx / CIN) % 512;
    int b = idx / (CIN * 512);
    int l = 512 + lp;
    const float* row = dec + ((size_t)(b * SL + l) * DM);
    float acc = projb[c];
    #pragma unroll 8
    for (int d = 0; d < DM; ++d) acc += row[d] * projW[d * CIN + c];
    out[idx] = sig[b * CIN + c] * (acc + mu[b * CIN + c]);
}

// ---------------------------------------------------------------------------
// host orchestration
// ---------------------------------------------------------------------------
static inline void gemm(const float* A, const float* W, const float* bias, float* C,
                        int M, int N, int K, int act, hipStream_t s)
{
    dim3 grid(N / 256, M / 64);
    gemm_wmma_kernel<<<grid, 256, 0, s>>>(A, W, bias, C, M, N, K, act);
}

extern "C" void kernel_launch(void* const* d_in, const int* in_sizes, int n_in,
                              void* d_out, int out_size, void* d_ws, size_t ws_size,
                              hipStream_t stream)
{
    (void)in_sizes; (void)n_in; (void)out_size; (void)ws_size;
    const float* P[80];
    for (int i = 0; i < 80; ++i) P[i] = (const float*)d_in[i];
    const float* x_enc = P[0];
    const float* x_mark_enc = P[1];
    const float* x_dec = P[2];
    const float* x_mark_dec = P[3];

    float* ws = (float*)d_ws;
    float* mu    = ws + 0;        // 64
    float* sig   = ws + 64;       // 64
    float* tau   = ws + 128;      // 16
    float* tauh  = ws + 256;      // 1024
    float* delh  = ws + 1280;     // 1024
    float* delta = ws + 2304;     // 8192
    const size_t BIG = (size_t)BB * SL * DM;             // 2 Mi floats
    float* A0 = ws + 16384;            // current x
    float* A1 = A0 + BIG;              // q / proj tmp
    float* A2 = A1 + BIG;              // k
    float* A3 = A2 + BIG;              // v
    float* A4 = A3 + BIG;              // attn-out / ffn-out
    float* A5 = A4 + BIG;              // saved x
    float* A6 = A5 + BIG;              // encoder output
    float* A7 = A6 + BIG;              // decoder intermediate
    float* H  = A7 + BIG;              // ffn hidden (8 Mi floats)

    const int M = BB * SL;                        // 8192 rows
    const dim3 attn_grid(SL / 64, NH, BB);

    // --- stationary factors ---
    stats_kernel<<<BB * CIN, 256, 0, stream>>>(x_enc, mu, sig);
    factor_hidden_kernel<<<dim3(DHID, BB), 256, 0, stream>>>(x_enc, sig, P[4], P[5], tauh);
    factor_hidden_kernel<<<dim3(DHID, BB), 256, 0, stream>>>(x_enc, mu,  P[8], P[9], delh);
    tau_kernel<<<BB, 128, 0, stream>>>(tauh, P[6], P[7], tau);
    delta_kernel<<<(BB * SL) / 256, 256, 0, stream>>>(delh, P[10], P[11], delta);

    // --- encoder ---
    embed_kernel<<<(unsigned)(BIG / 256), 256, 0, stream>>>(
        x_enc, x_mark_enc, P[12], P[13], mu, sig, A0, 1);
    for (int layer = 0; layer < 2; ++layer) {
        const float* const* L = P + 16 + 16 * layer;
        gemm(A0, L[0], L[1], A1, M, DM, DM, 0, stream);       // Q
        gemm(A0, L[2], L[3], A2, M, DM, DM, 0, stream);       // K
        gemm(A0, L[4], L[5], A3, M, DM, DM, 0, stream);       // V
        attn_wmma_kernel<<<attn_grid, 128, 0, stream>>>(A1, A2, A3, A4, tau, delta, 0, 1);
        gemm(A4, L[6], L[7], A1, M, DM, DM, 0, stream);       // Wo
        add_ln_kernel<<<M, 256, 0, stream>>>(A0, A1, L[12], L[13], A5);   // ln1
        gemm(A5, L[8], L[9], H, M, DFF, DM, 1, stream);       // gelu(x W1 + b1)
        gemm(H, L[10], L[11], A4, M, DM, DFF, 0, stream);     // W2
        add_ln_kernel<<<M, 256, 0, stream>>>(A5, A4, L[14], L[15], A0);   // ln2
    }
    add_ln_kernel<<<M, 256, 0, stream>>>(A0, nullptr, P[48], P[49], A6);  // enc final LN

    // --- decoder ---
    embed_kernel<<<(unsigned)(BIG / 256), 256, 0, stream>>>(
        x_dec, x_mark_dec, P[14], P[15], mu, sig, A0, 0);
    {
        const float* const* L = P + 50;
        // self attention (causal, no delta)
        gemm(A0, L[0], L[1], A1, M, DM, DM, 0, stream);
        gemm(A0, L[2], L[3], A2, M, DM, DM, 0, stream);
        gemm(A0, L[4], L[5], A3, M, DM, DM, 0, stream);
        attn_wmma_kernel<<<attn_grid, 128, 0, stream>>>(A1, A2, A3, A4, tau, delta, 1, 0);
        gemm(A4, L[6], L[7], A1, M, DM, DM, 0, stream);
        add_ln_kernel<<<M, 256, 0, stream>>>(A0, A1, L[20], L[21], A5);   // ln1
        // cross attention (delta, no mask); K/V from encoder output
        gemm(A5, L[8],  L[9],  A1, M, DM, DM, 0, stream);
        gemm(A6, L[10], L[11], A2, M, DM, DM, 0, stream);
        gemm(A6, L[12], L[13], A3, M, DM, DM, 0, stream);
        attn_wmma_kernel<<<attn_grid, 128, 0, stream>>>(A1, A2, A3, A4, tau, delta, 0, 1);
        gemm(A4, L[14], L[15], A1, M, DM, DM, 0, stream);
        add_ln_kernel<<<M, 256, 0, stream>>>(A5, A1, L[22], L[23], A7);   // ln2
        // FFN
        gemm(A7, L[16], L[17], H, M, DFF, DM, 1, stream);
        gemm(H, L[18], L[19], A4, M, DM, DFF, 0, stream);
        add_ln_kernel<<<M, 256, 0, stream>>>(A7, A4, L[24], L[25], A0);   // ln3
    }
    add_ln_kernel<<<M, 256, 0, stream>>>(A0, nullptr, P[76], P[77], A5);  // dec final LN

    // --- projection + de-normalization + slice last 512 ---
    output_kernel<<<(BB * 512 * CIN + 255) / 256, 256, 0, stream>>>(
        A5, P[78], P[79], mu, sig, (float*)d_out);
}
